// Graph_Module_9801115369537
// MI455X (gfx1250) — matrix operand order repaired and checked by reading, not compiled, not run on hardware
//
#include <hip/hip_runtime.h>

// ---------------------------------------------------------------------------
// MI455X / gfx1250: wave32, WMMA f32_16x16x32_f16 (f32 accumulate),
// double-buffered LDS fed by GLOBAL_LOAD_ASYNC_TO_LDS_B128 (ASYNCcnt).
// Pipeline (all GEMMs via one templated WMMA kernel):
//   S0  adj cast   : d_out[0:83886080) = float(graph)
//   P   prep       : X->f16, W transposes -> f16 (N-major "BT" layout)
//   S1  sup1T[b]   = (X_b @ Wg1_allheads)^T            f16   M1024 N1024 K256
//   S2  h1[b]      = relu(Adj_t @ sup1 + bg1)          f16   M1024 N512  K1024  (x2 t)
//   S3  sup2T[b,h] = (h1_h @ Wg2_h)^T                  f16   M1024 N64   K256   (x4 h)
//   S4  g[b]       = Adj_t @ sup2 + bg2                f32   M1024 N128  K1024  (x2 t)
//   S5  LayerNorm(ddof=1, a*(x-mu)/(sd+eps)+b) + residual; emit f32 + f16
//   S6  ff1 = relu(g @ Wf1 + bf1)                      f16   M16384 N256 K256
//   S7  out = ff1 @ Wf2 + bf2 + g                      f32   M16384 N256 K256
// ---------------------------------------------------------------------------

typedef __attribute__((ext_vector_type(16))) _Float16 v16h;
typedef __attribute__((ext_vector_type(8)))  _Float16 v8h;
typedef __attribute__((ext_vector_type(4)))  _Float16 v4h;
typedef __attribute__((ext_vector_type(8)))  float    v8f;
typedef __attribute__((ext_vector_type(4)))  float    v4f;
typedef __attribute__((ext_vector_type(4)))  int      v4i;

enum { MODE_TRANS_F16 = 0, MODE_RELU_F16 = 1, MODE_BIAS_F32 = 2, MODE_BIAS_RES_F32 = 3 };

// Low 32 bits of a generic pointer to LDS are the LDS byte address (ISA 10.2:
// the LDS address is its low 32 bits) -- used as the VDST operand of async-to-LDS ops.
__device__ __forceinline__ unsigned ldsAddr(const void* p) {
  return (unsigned)(unsigned long long)p;
}

__device__ __forceinline__ void asyncLoadB128(unsigned lds, const void* gaddr) {
  asm volatile("global_load_async_to_lds_b128 %0, %1, off"
               :: "v"(lds), "v"((unsigned long long)gaddr) : "memory");
}

// ---------------------------------------------------------------------------
// Generic tiled WMMA GEMM:  C(MxN) = A(MxK) @ B(KxN) [+bias, +res, relu, ^T]
// B is stored N-major ("BT", rows of length K) so the B fragment is a
// pair of 8xf16 per-lane loads (K = 8*hi.., 16+8*hi..) matching the 16-bit B VGPR layout.
// Block: 256 thr = 8 waves (4x2), macro tile 128x64, wave tile 32x32 (2x2 frags).
// AINT=1: A elements are int32 (adjacency), converted to f16 while staging LDS
//         (async DMA cannot convert, so that path stays through VGPRs).
// ---------------------------------------------------------------------------
template <int MODE, int AINT>
__global__ __launch_bounds__(256) void gemm_wmma(
    const void* __restrict__ Abase, int lda, long AsI, long AsO,
    const _Float16* __restrict__ Btbase, int ldb, long BsI, long BsO,
    void* __restrict__ Cbase, int ldc, long CsI, long CsO,
    const float* __restrict__ bias, int biasStride,
    const float* __restrict__ res,
    int Kdim, int nInner)
{
  __shared__ _Float16 lA[2][128 * 32];
  __shared__ _Float16 lB[2][64 * 32];

  const int z     = blockIdx.z;
  const int inner = z % nInner;
  const int outer = z / nInner;
  const int tid   = threadIdx.x;
  const int wid   = tid >> 5;
  const int lane  = tid & 31;
  const int hi    = lane >> 4;   // K-half selector in 16-bit WMMA layouts
  const int l16   = lane & 15;
  const int waveM = wid & 3;
  const int waveN = wid >> 2;

  const long aOff = (long)inner * AsI + (long)outer * AsO;
  const _Float16* Ah = (const _Float16*)Abase + aOff;
  const int*      Ai = (const int*)Abase + aOff;
  const _Float16* Bt = Btbase + (long)inner * BsI + (long)outer * BsO;

  const int m0 = blockIdx.x * 128;
  const int n0 = blockIdx.y * 64;

  // staging split: A tile 128x32 (16 elems/thread), B tile 64x32 (8/thread)
  const int arow = tid >> 1;
  const int acol = (tid & 1) * 16;
  const int brow = tid >> 2;
  const int bcol = (tid & 3) * 8;

  v8f c[2][2] = {};

  // Issue one k-tile of A+B into LDS buffer `buf`.
  auto issueTile = [&](int kk, int buf) {
    // B tile: async DMA, 16 B per lane
    asyncLoadB128(ldsAddr(&lB[buf][brow * 32 + bcol]),
                  Bt + (long)(n0 + brow) * ldb + kk + bcol);
    if (AINT) {
      // int32 adjacency -> f16 while staging (one conversion per element)
      const int* src = Ai + (long)(m0 + arow) * lda + kk + acol;
#pragma unroll
      for (int t = 0; t < 2; ++t) {
        v4i xa = *(const v4i*)(src + t * 8);
        v4i xb = *(const v4i*)(src + t * 8 + 4);
        v8h p;
        p[0] = (_Float16)(float)xa[0]; p[1] = (_Float16)(float)xa[1];
        p[2] = (_Float16)(float)xa[2]; p[3] = (_Float16)(float)xa[3];
        p[4] = (_Float16)(float)xb[0]; p[5] = (_Float16)(float)xb[1];
        p[6] = (_Float16)(float)xb[2]; p[7] = (_Float16)(float)xb[3];
        *(v8h*)(&lA[buf][arow * 32 + acol + t * 8]) = p;
      }
      if (kk + 32 < Kdim)  // prefetch the tile after this one
        __builtin_prefetch((const void*)(src + 32), 0, 0);
    } else {
      const _Float16* src = Ah + (long)(m0 + arow) * lda + kk + acol;
      unsigned la = ldsAddr(&lA[buf][arow * 32 + acol]);
      asyncLoadB128(la, src);
      asyncLoadB128(la + 16u, src + 8);
    }
  };

  const int nsteps = Kdim >> 5;
  issueTile(0, 0);
  for (int s = 0; s < nsteps; ++s) {
    const int cur   = s & 1;
    const bool more = (s + 1) < nsteps;
    if (more) issueTile((s + 1) << 5, cur ^ 1);

    // Per-wave async groups complete in order: waiting <= size of the
    // just-issued group guarantees the current tile's group has landed.
    if (more) {
      if (AINT) asm volatile("s_wait_asynccnt 0x1" ::: "memory");
      else      asm volatile("s_wait_asynccnt 0x3" ::: "memory");
    } else {
      asm volatile("s_wait_asynccnt 0x0" ::: "memory");
    }
    __syncthreads();

    // ---- gather fragments (per documented 16-bit VGPR layouts) ----
    v16h a[2], b[2];
#pragma unroll
    for (int i = 0; i < 2; ++i) {
      const _Float16* ar = &lA[cur][(waveM * 32 + i * 16 + l16) * 32];
      v8h a0 = *(const v8h*)(ar + hi * 8);        // K = hi*8 .. +7
      v8h a1 = *(const v8h*)(ar + 16 + hi * 8);   // K = 16+hi*8 .. +7
      a[i] = __builtin_shufflevector(a0, a1, 0, 1, 2, 3, 4, 5, 6, 7,
                                     8, 9, 10, 11, 12, 13, 14, 15);
    }
#pragma unroll
    for (int j = 0; j < 2; ++j) {
      const _Float16* br = &lB[cur][(waveN * 32 + j * 16 + l16) * 32];
      v8h b0 = *(const v8h*)(br + hi * 8);
      v8h b1 = *(const v8h*)(br + 16 + hi * 8);
      b[j] = __builtin_shufflevector(b0, b1, 0, 1, 2, 3, 4, 5, 6, 7,
                                     8, 9, 10, 11, 12, 13, 14, 15);
    }
#pragma unroll
    for (int i = 0; i < 2; ++i)
#pragma unroll
      for (int j = 0; j < 2; ++j)
        c[i][j] = __builtin_amdgcn_wmma_f32_16x16x32_f16(
            false, a[i], false, b[j], (short)0, c[i][j], false, false);
    __syncthreads();  // everyone done reading buf `cur` before it is refilled
  }

  // ---- epilogue ----
#pragma unroll
  for (int i = 0; i < 2; ++i) {
#pragma unroll
    for (int j = 0; j < 2; ++j) {
      const int gn  = n0 + waveN * 32 + j * 16 + l16;       // column
      const int gm0 = m0 + waveM * 32 + i * 16 + hi * 8;    // rows gm0..gm0+7
      if (MODE == MODE_TRANS_F16) {
        _Float16* C = (_Float16*)Cbase + (long)inner * CsI + (long)outer * CsO;
        v8h p;
#pragma unroll
        for (int v = 0; v < 8; ++v) p[v] = (_Float16)c[i][j][v];
        *(v8h*)(&C[(long)gn * ldc + gm0]) = p;              // ldc = M extent
      } else if (MODE == MODE_RELU_F16) {
        _Float16* C = (_Float16*)Cbase + (long)inner * CsI + (long)outer * CsO;
        const float bia = bias[outer * biasStride + gn];
#pragma unroll
        for (int v = 0; v < 8; ++v) {
          float x = c[i][j][v] + bia;
          x = x > 0.f ? x : 0.f;
          C[(long)(gm0 + v) * ldc + gn] = (_Float16)x;
        }
      } else if (MODE == MODE_BIAS_F32) {
        float* C = (float*)Cbase + (long)inner * CsI + (long)outer * CsO;
        const float bia = bias[outer * biasStride + gn];
#pragma unroll
        for (int v = 0; v < 8; ++v)
          C[(long)(gm0 + v) * ldc + gn] = c[i][j][v] + bia;
      } else {  // MODE_BIAS_RES_F32
        float* C = (float*)Cbase + (long)inner * CsI + (long)outer * CsO;
        const float bia = bias[outer * biasStride + gn];
#pragma unroll
        for (int v = 0; v < 8; ++v) {
          const long idx = (long)(gm0 + v) * ldc + gn;
          C[idx] = c[i][j][v] + bia + res[idx];
        }
      }
    }
  }
}

// int32 -> float cast of the whole graph tensor (tuple output element 0)
__global__ __launch_bounds__(256) void k_cast_adj(const int* __restrict__ gin,
                                                  float* __restrict__ out, long n4)
{
  const long i = (long)blockIdx.x * 256 + threadIdx.x;
  if (i < n4) {
    v4i x = *(const v4i*)(gin + i * 4);
    v4f y;
    y[0] = (float)x[0]; y[1] = (float)x[1]; y[2] = (float)x[2]; y[3] = (float)x[3];
    *(v4f*)(out + i * 4) = y;
  }
}

// f32 -> f16 cast, 4 elements/thread
__global__ __launch_bounds__(256) void k_cast_f16x4(const float* __restrict__ in,
                                                    _Float16* __restrict__ out, long n4)
{
  const long i = (long)blockIdx.x * 256 + threadIdx.x;
  if (i < n4) {
    v4f x = *(const v4f*)(in + i * 4);
    v4h p;
    p[0] = (_Float16)x[0]; p[1] = (_Float16)x[1];
    p[2] = (_Float16)x[2]; p[3] = (_Float16)x[3];
    *(v4h*)(out + i * 4) = p;
  }
}

// batched (rows x cols) f32 -> (cols x rows) f16 transpose (small weights)
__global__ __launch_bounds__(256) void k_transpose_f16(const float* __restrict__ in,
                                                       _Float16* __restrict__ out,
                                                       int rows, int cols)
{
  const long base = (long)blockIdx.y * rows * cols;
  const int idx = blockIdx.x * 256 + threadIdx.x;
  if (idx < rows * cols) {
    const int r = idx / cols, cc = idx % cols;
    out[base + (long)cc * rows + r] = (_Float16)in[base + idx];
  }
}

// LayerNorm (MWPToolkit: a*(x-mu)/(std_unbiased+eps)+b) + residual; one row/block
__global__ __launch_bounds__(256) void k_layernorm(float* __restrict__ g,
                                                   const float* __restrict__ xin,
                                                   const float* __restrict__ ln_a,
                                                   const float* __restrict__ ln_b,
                                                   _Float16* __restrict__ gH)
{
  const long row = blockIdx.x;
  const int  t   = threadIdx.x;
  const float x  = g[row * 256 + t];

  float s = x, ss = x * x;
#pragma unroll
  for (int o = 16; o > 0; o >>= 1) {
    s  += __shfl_down(s, o, 32);
    ss += __shfl_down(ss, o, 32);
  }
  __shared__ float sA[8], sB[8];
  if ((t & 31) == 0) { sA[t >> 5] = s; sB[t >> 5] = ss; }
  __syncthreads();
  float S = 0.f, SS = 0.f;
#pragma unroll
  for (int i = 0; i < 8; ++i) { S += sA[i]; SS += sB[i]; }

  const float mu  = S * (1.0f / 256.0f);
  float var = (SS - 256.0f * mu * mu) * (1.0f / 255.0f);  // ddof=1
  var = var > 0.f ? var : 0.f;
  const float sd = sqrtf(var);
  const float y  = ln_a[t] * (x - mu) / (sd + 1e-6f) + ln_b[t] + xin[row * 256 + t];
  g[row * 256 + t]  = y;
  gH[row * 256 + t] = (_Float16)y;
}

extern "C" void kernel_launch(void* const* d_in, const int* in_sizes, int n_in,
                              void* d_out, int out_size, void* d_ws, size_t ws_size,
                              hipStream_t stream)
{
  (void)in_sizes; (void)n_in; (void)out_size; (void)ws_size;

  const float* gnodes = (const float*)d_in[0];   // (16,1024,256)
  const int*   graph  = (const int*)d_in[1];     // (16,5,1024,1024)
  const float* Wg1 = (const float*)d_in[2];      // (4,256,256)
  const float* bg1 = (const float*)d_in[3];      // (4,256)
  const float* Wg2 = (const float*)d_in[4];      // (4,256,64)
  const float* bg2 = (const float*)d_in[5];      // (4,64)
  const float* Wf1 = (const float*)d_in[6];      // (256,256)
  const float* bf1 = (const float*)d_in[7];      // (256,)
  const float* Wf2 = (const float*)d_in[8];      // (256,256)
  const float* bf2 = (const float*)d_in[9];      // (256,)
  const float* ln_a = (const float*)d_in[10];
  const float* ln_b = (const float*)d_in[11];

  float* outAdj = (float*)d_out;                          // 83,886,080 floats
  float* outF   = outAdj + (long)16 * 5 * 1024 * 1024;    // 4,194,304 floats

  // workspace layout (96 MiB, regions overlaid where lifetimes allow)
  char* ws = (char*)d_ws;
  _Float16* Xb    = (_Float16*)(ws);                                   // 8 MiB
  _Float16* Wg1T  = (_Float16*)(ws + (8l << 20));                      // 512 KiB (4,256,256)
  _Float16* Wg2T  = (_Float16*)(ws + (8l << 20) + (1l << 19));         // 128 KiB (4,64,256)
  _Float16* Wf1T  = (_Float16*)(ws + (8l << 20) + (1l << 19) + (1l << 17));
  _Float16* Wf2T  = (_Float16*)(ws + (8l << 20) + (1l << 19) + (2l << 17));
  _Float16* sup1T = (_Float16*)(ws + (16l << 20));   // 32 MiB (b,1024,1024)
  _Float16* sup2T = sup1T;                           // overlay after S2 (b,h,64,1024)
  _Float16* h1H   = (_Float16*)(ws + (48l << 20));   // 32 MiB (b,1024,1024)
  _Float16* gH    = h1H;                             // overlay after S3 (16384,256)
  _Float16* ff1H  = (_Float16*)(ws + (56l << 20));   // 8 MiB (16384,256)
  float*    g     = (float*)(ws + (80l << 20));      // 16 MiB (16384,256)

  // S0: tuple output element 0 = float(graph)
  k_cast_adj<<<81920, 256, 0, stream>>>(graph, outAdj, 20971520l);

  // Prep: casts + weight transposes into the N-major (BT) layouts
  k_cast_f16x4<<<4096, 256, 0, stream>>>(gnodes, Xb, 1048576l);
  k_transpose_f16<<<dim3(256, 4), 256, 0, stream>>>(Wg1, Wg1T, 256, 256);
  k_transpose_f16<<<dim3(64, 4), 256, 0, stream>>>(Wg2, Wg2T, 256, 64);
  k_transpose_f16<<<dim3(256, 1), 256, 0, stream>>>(Wf1, Wf1T, 256, 256);
  k_transpose_f16<<<dim3(256, 1), 256, 0, stream>>>(Wf2, Wf2T, 256, 256);

  // S1: sup1T[b] = (X_b @ Wg1_allheads)^T   (per b; N = h*256+f)
  gemm_wmma<MODE_TRANS_F16, 0><<<dim3(8, 16, 16), 256, 0, stream>>>(
      Xb, 256, 262144, 0,  Wg1T, 256, 0, 0,
      sup1T, 1024, 1048576, 0,  nullptr, 0, nullptr, 256, 16);

  // S2: h1[b][:, t*512..] = relu(Adj(c=1 or 4) @ sup1 + bg1) -> f16 row-major
  //     A = graph slice (int32, converted on the fly): base c=1, +3*K*K for c=4
  gemm_wmma<MODE_RELU_F16, 1><<<dim3(8, 8, 32), 256, 0, stream>>>(
      graph + 1048576, 1024, 5242880, 3145728,
      sup1T, 1024, 1048576, 524288,
      h1H, 1024, 1048576, 512,  bg1, 512, nullptr, 1024, 16);

  // S3: sup2T[b][h] = (h1_h @ Wg2_h)^T
  gemm_wmma<MODE_TRANS_F16, 0><<<dim3(8, 1, 64), 256, 0, stream>>>(
      h1H, 1024, 1048576, 256,  Wg2T, 256, 0, 16384,
      sup2T, 1024, 262144, 65536,  nullptr, 0, nullptr, 256, 16);

  // S4: g[b][:, t*128..] = Adj @ sup2(2 heads) + bg2  -> f32 (heads concat)
  gemm_wmma<MODE_BIAS_F32, 1><<<dim3(8, 2, 32), 256, 0, stream>>>(
      graph + 1048576, 1024, 5242880, 3145728,
      sup2T, 1024, 262144, 131072,
      g, 256, 262144, 128,  bg2, 128, nullptr, 1024, 16);

  // S5: LayerNorm + residual (in place on g), f16 copy for the FFN
  k_layernorm<<<16384, 256, 0, stream>>>(g, gnodes, ln_a, ln_b, gH);

  // S6: ff1 = relu(g @ Wf1 + bf1)   (all batches fused: M = 16384)
  gemm_wmma<MODE_RELU_F16, 0><<<dim3(128, 4, 1), 256, 0, stream>>>(
      gH, 256, 0, 0,  Wf1T, 256, 0, 0,
      ff1H, 256, 0, 0,  bf1, 0, nullptr, 256, 1);

  // S7: out = ff1 @ Wf2 + bf2 + g
  gemm_wmma<MODE_BIAS_RES_F32, 0><<<dim3(128, 4, 1), 256, 0, stream>>>(
      ff1H, 256, 0, 0,  Wf2T, 256, 0, 0,
      outF, 256, 0, 0,  bf2, 0, g, 256, 1);
}
